// wFMConv2d_27676769255551
// MI455X (gfx1250) — matrix operand order repaired and checked by reading, not compiled, not run on hardware
//
#include <hip/hip_runtime.h>

// ---------------------------------------------------------------------------
// Problem constants (from reference): B=16, C=64, H=W=64, K=3x3, Cout=64
// L = 64*64 = 4096, N = B*L = 65536
// ---------------------------------------------------------------------------
#define BB   16
#define CC_  64
#define HH   64
#define WW   64
#define COUT 64
#define LL   4096
#define NN   65536
#define EPSF 1e-5f

typedef __attribute__((ext_vector_type(16))) __bf16 v16bf;
typedef __attribute__((ext_vector_type(8)))  float  v8f;

__device__ __forceinline__ unsigned short f2bf(float f) {
    unsigned int u = __float_as_uint(f);
    unsigned int r = u + 0x7FFFu + ((u >> 16) & 1u);   // round-to-nearest-even
    return (unsigned short)(r >> 16);
}

__device__ __forceinline__ float blockReduceSum(float v, float* sbuf) {
    int t = threadIdx.x;
    sbuf[t] = v; __syncthreads();
    for (int s = 128; s > 0; s >>= 1) {
        if (t < s) sbuf[t] += sbuf[t + s];
        __syncthreads();
    }
    float r = sbuf[0]; __syncthreads();
    return r;
}

// ---------------------------------------------------------------------------
// Kernel 0: weight normalizations + pre-swizzle wmm2/wma2 into WMMA A-layout.
// A-layout (16x32 bf16, contiguous k order, same as B): element e of lane ln (v16bf) holds
//   M = 16*mt + (ln&15),  K = 32*kt + ((ln>=16)?16:0) + e
// stored flat as [((mt*2+kt)*32 + ln)*16 + e]  -> per-lane 32B contiguous.
// ---------------------------------------------------------------------------
__global__ void wfm_prep(const float* __restrict__ w_mag,
                         const float* __restrict__ w_ang,
                         const float* __restrict__ w1,
                         const float* __restrict__ w2,
                         float* __restrict__ nws, float* __restrict__ nwa,
                         float* __restrict__ wm1, float* __restrict__ wa1,
                         unsigned short* __restrict__ Am,
                         unsigned short* __restrict__ Aa) {
    __shared__ float sbuf[256];
    __shared__ float row1[64];
    __shared__ float row2[64];
    int t = threadIdx.x;

    float pm = 0.f, pa = 0.f, p1 = 0.f;
    for (int i = t; i < CC_ * 9; i += 256) {
        float a = w_mag[i]; pm += a * a;
        float b = w_ang[i]; pa += b * b;
        float c = w1[i];    p1 += c * c;
    }
    float s_mag = blockReduceSum(pm, sbuf);
    float s_ang = blockReduceSum(pa, sbuf);
    float s_w1  = blockReduceSum(p1, sbuf);

    if (t < 64) {
        float r = 0.f;
        for (int k = 0; k < 9; ++k) { float v = w1[t * 9 + k]; r += v * v; }
        row1[t] = r;
        float r2 = 0.f;
        for (int c = 0; c < CC_; ++c) { float v = w2[t * CC_ + c]; r2 += v * v; }
        row2[t] = r2;
    }
    __syncthreads();
    float s_w2 = blockReduceSum((t < 64) ? row2[t] : 0.f, sbuf);

    for (int i = t; i < CC_ * 9; i += 256) {
        int c = i / 9;
        float a = w_mag[i]; nws[i] = a * a / s_mag;
        float b = w_ang[i]; nwa[i] = b / s_ang;
        float v = w1[i];    wm1[i] = v * v / row1[c];  wa1[i] = v * v / s_w1;
    }

    // A-operand swizzle for wmm2 (mag) and wma2 (ang)
    for (int idx = t; idx < 4096; idx += 256) {
        int e    = idx & 15;
        int lane = (idx >> 4) & 31;
        int kt   = (idx >> 9) & 1;
        int mt   = (idx >> 10) & 3;
        int o = mt * 16 + (lane & 15);
        int k = kt * 32 + ((lane >> 4) & 1) * 16 + e;
        float v  = w2[o * CC_ + k];
        float v2 = v * v;
        Am[idx] = f2bf(v2 / row2[o]);
        Aa[idx] = f2bf(v2 / s_w2);
    }
}

// ---------------------------------------------------------------------------
// Kernel 1: fused fold/unfold collapse + depthwise 3x3, transposed bf16 store.
// Block = 256 thr; covers one (b, h) row chunk of 16 pixels x all 64 channels.
// Output layout m1T/a1T: [n][c] with n = b*4096 + h*64 + w  (B-operand ready).
// ---------------------------------------------------------------------------
__global__ void __launch_bounds__(256, 1)
wfm_stage1(const float* __restrict__ x_mag,
           const float* __restrict__ x_ang,
           const float* __restrict__ nws,
           const float* __restrict__ nwa,
           const float* __restrict__ wm1,
           const float* __restrict__ wa1,
           unsigned short* __restrict__ m1T,
           unsigned short* __restrict__ a1T) {
    __shared__ unsigned short lm[16][72];
    __shared__ unsigned short la[16][72];
    const float LOG_EPS = -11.512925464970229f;   // log(1e-5)

    int bid = blockIdx.x;
    int b   = bid >> 8;            // 256 blocks per batch (64 h * 4 wgroups)
    int rem = bid & 255;
    int h   = rem >> 2;
    int w0  = (rem & 3) << 4;

    int t  = threadIdx.x;
    int wl = t & 15;               // pixel within chunk (coalesced reads)
    int cg = t >> 4;
    int w  = w0 + wl;

    for (int ccb = 0; ccb < 4; ++ccb) {
        int c = cg + (ccb << 4);
        float ws9[9], wa9[9], q1[9], q2[9];
        float Sb = 0.f, Sa = 0.f;
        #pragma unroll
        for (int k = 0; k < 9; ++k) {
            ws9[k] = nws[c * 9 + k];
            wa9[k] = nwa[c * 9 + k];
            q1[k]  = wm1[c * 9 + k];
            q2[k]  = wa1[c * 9 + k];
            Sb += ws9[k]; Sa += wa9[k];
        }
        const float* xm = x_mag + (size_t)(b * CC_ + c) * LL;
        const float* xa = x_ang + (size_t)(b * CC_ + c) * LL;

        float accm = 0.f, acca = 0.f;
        #pragma unroll
        for (int kh = 0; kh < 3; ++kh) {
            int y = h + kh - 1;
            #pragma unroll
            for (int kw = 0; kw < 3; ++kw) {
                int x = w + kw - 1;
                float wk1 = q1[kh * 3 + kw];
                float wk2 = q2[kh * 3 + kw];
                if (y >= 0 && y < HH && x >= 0 && x < WW) {
                    int cy = 1 + (y > 0) + (y < HH - 1);
                    int cx = 1 + (x > 0) + (x < WW - 1);
                    float bias, abias;
                    if (y >= 1 && y <= HH - 2 && x >= 1 && x <= WW - 2) {
                        bias = Sb; abias = Sa;        // interior fast path
                    } else {
                        int khlo = (y >= HH - 1) ? 1 : 0, khhi = (y <= 0) ? 1 : 2;
                        int kwlo = (x >= WW - 1) ? 1 : 0, kwhi = (x <= 0) ? 1 : 2;
                        bias = 0.f; abias = 0.f;
                        for (int i = khlo; i <= khhi; ++i)
                            for (int j = kwlo; j <= kwhi; ++j) {
                                bias  += ws9[i * 3 + j];
                                abias += wa9[i * 3 + j];
                            }
                    }
                    float fmv = xm[y * WW + x] * (float)(cy * cx) + bias;
                    accm += wk1 * __logf(fmv + EPSF);
                    acca += wk2 * (xa[y * WW + x] * abias);
                } else {
                    accm += wk1 * LOG_EPS;            // log(0 + EPS) from zero pad
                }
            }
        }
        lm[wl][c] = f2bf(accm);
        la[wl][c] = f2bf(acca);
    }
    __syncthreads();

    size_t n0 = (size_t)b * LL + (size_t)h * WW + w0;
    for (int f = t; f < 1024; f += 256) {            // coalesced transposed store
        int wl2 = f >> 6;
        int c2  = f & 63;
        size_t off = (n0 + wl2) * CC_ + c2;
        m1T[off] = lm[wl2][c2];
        a1T[off] = la[wl2][c2];
    }
}

// ---------------------------------------------------------------------------
// Kernel 2: channel-mixing GEMM via v_wmma_f32_16x16x32_bf16 + exp/sincos.
// Each wave: 16 output columns, full M=64, both mag & ang -> 16 WMMAs.
// __launch_bounds__(256,1): 8 waves/block, grid=512 -> occupancy is not the
// limit; give the compiler the full VGPR file so the 8 accumulator tiles +
// B operands stay in registers (no scratch spills).
// ---------------------------------------------------------------------------
__global__ void __launch_bounds__(256, 1)
wfm_stage2(const unsigned short* __restrict__ m1T,
           const unsigned short* __restrict__ a1T,
           const unsigned short* __restrict__ Am,
           const unsigned short* __restrict__ Aa,
           float* __restrict__ out) {
    int lane = threadIdx.x & 31;
    int wid  = (int)((blockIdx.x * blockDim.x + threadIdx.x) >> 5);
    size_t nb = (size_t)wid * 16;

    // gfx1250 prefetch of the (tiny, shared) A operand tables
    __builtin_prefetch(Am + lane * 16, 0, 0);
    __builtin_prefetch(Aa + lane * 16, 0, 0);

    // B operand (contiguous k order, matching A): lane holds column n = nb + (lane&15),
    // K rows kbase..kbase+15 with kbase = kt*32 + ((lane>=16)?16:0).
    size_t ncol  = nb + (lane & 15);
    int    kbhi  = ((lane >> 4) & 1) * 16;
    const unsigned short* brm = m1T + ncol * CC_ + kbhi;
    const unsigned short* bra = a1T + ncol * CC_ + kbhi;
    v16bf Bm0 = *reinterpret_cast<const v16bf*>(brm);
    v16bf Bm1 = *reinterpret_cast<const v16bf*>(brm + 32);
    v16bf Ba0 = *reinterpret_cast<const v16bf*>(bra);
    v16bf Ba1 = *reinterpret_cast<const v16bf*>(bra + 32);

    int b     = (int)(nb >> 12);                 // 4096 % 16 == 0 -> single batch
    int lbase = (int)(nb & (LL - 1)) + (lane & 15);
    int ohalf = ((lane >> 4) & 1) * 8;           // C/D: lanes 16-31 -> M+8

    #pragma unroll
    for (int mt = 0; mt < 4; ++mt) {
        v16bf Am0 = *reinterpret_cast<const v16bf*>(Am + ((mt * 2 + 0) * 32 + lane) * 16);
        v16bf Am1 = *reinterpret_cast<const v16bf*>(Am + ((mt * 2 + 1) * 32 + lane) * 16);
        v16bf Aa0 = *reinterpret_cast<const v16bf*>(Aa + ((mt * 2 + 0) * 32 + lane) * 16);
        v16bf Aa1 = *reinterpret_cast<const v16bf*>(Aa + ((mt * 2 + 1) * 32 + lane) * 16);

        v8f cm = {};
        v8f ca = {};
        cm = __builtin_amdgcn_wmma_f32_16x16x32_bf16(false, Am0, false, Bm0, (short)0, cm, false, false);
        cm = __builtin_amdgcn_wmma_f32_16x16x32_bf16(false, Am1, false, Bm1, (short)0, cm, false, false);
        ca = __builtin_amdgcn_wmma_f32_16x16x32_bf16(false, Aa0, false, Ba0, (short)0, ca, false, false);
        ca = __builtin_amdgcn_wmma_f32_16x16x32_bf16(false, Aa1, false, Ba1, (short)0, ca, false, false);

        #pragma unroll
        for (int v = 0; v < 8; ++v) {
            int o = mt * 16 + ohalf + v;
            float mag = __expf(cm[v]);
            float ang = ca[v];
            float sn, cs;
            __sincosf(ang, &sn, &cs);
            out[(((size_t)b * 2 + 0) * COUT + o) * LL + lbase] = mag * cs;
            out[(((size_t)b * 2 + 1) * COUT + o) * LL + lbase] = mag * sn;
        }
    }
}

// ---------------------------------------------------------------------------
// Host entry
// ---------------------------------------------------------------------------
extern "C" void kernel_launch(void* const* d_in, const int* in_sizes, int n_in,
                              void* d_out, int out_size, void* d_ws, size_t ws_size,
                              hipStream_t stream) {
    (void)in_sizes; (void)n_in; (void)out_size; (void)ws_size;
    const float* x_mag = (const float*)d_in[0];
    const float* x_ang = (const float*)d_in[1];
    const float* w_mag = (const float*)d_in[2];
    const float* w_ang = (const float*)d_in[3];
    const float* w1    = (const float*)d_in[4];
    const float* w2    = (const float*)d_in[5];
    float* out = (float*)d_out;

    // workspace carve (~16.8 MB total)
    char* ws = (char*)d_ws;
    float* nws = (float*)ws;            // 576 f32
    float* nwa = nws + 576;
    float* wm1 = nwa + 576;
    float* wa1 = wm1 + 576;
    unsigned short* Am = (unsigned short*)(wa1 + 576);  // 4096 bf16
    unsigned short* Aa = Am + 4096;
    size_t base = ((size_t)((char*)(Aa + 4096) - ws) + 255) & ~(size_t)255;
    unsigned short* m1T = (unsigned short*)(ws + base);        // NN*64 bf16 (8 MB)
    unsigned short* a1T = m1T + (size_t)NN * CC_;              // 8 MB

    wfm_prep<<<1, 256, 0, stream>>>(w_mag, w_ang, w1, w2, nws, nwa, wm1, wa1, Am, Aa);
    wfm_stage1<<<BB * HH * (WW / 16), 256, 0, stream>>>(x_mag, x_ang, nws, nwa, wm1, wa1, m1T, a1T);
    wfm_stage2<<<(NN / 16) / 8, 256, 0, stream>>>(m1T, a1T, Am, Aa, out);
}
